// TransformerBlock_69174743269908
// MI455X (gfx1250) — matrix operand order repaired and checked by reading, not compiled, not run on hardware
//
#include <hip/hip_runtime.h>
#include <cstdint>
#include <cstddef>

typedef __attribute__((ext_vector_type(16))) __bf16 v16bf;
typedef __attribute__((ext_vector_type(8)))  __bf16 v8bf;
typedef __attribute__((ext_vector_type(8)))  float  v8f;
typedef __attribute__((ext_vector_type(4)))  unsigned int v4u;
typedef __attribute__((ext_vector_type(4)))  int v4i;
typedef __attribute__((ext_vector_type(8)))  int v8i;

#define BB    4
#define TT    2048
#define CC    768
#define HH    12
#define DD    64
#define FFD   3072
#define QKVLD 2304          // 3*C row stride of qkv buffer
#define ROWS  (BB * TT)     // 8192

// ---------- helpers ----------
__device__ __forceinline__ unsigned short f2bf(float f) {
    union { float f; unsigned u; } c; c.f = f;
    unsigned r = c.u + 0x7FFFu + ((c.u >> 16) & 1u);   // round-to-nearest-even
    return (unsigned short)(r >> 16);
}
__device__ __forceinline__ __bf16 us2bf(unsigned short u) {
    union { unsigned short u; __bf16 b; } c; c.u = u;
    return c.b;
}
__device__ __forceinline__ v8f zero8() {
    v8f v;
#pragma unroll
    for (int i = 0; i < 8; ++i) v[i] = 0.0f;
    return v;
}
union BF16x16 { v16bf v; v8bf h[2]; };
// fragment = two contiguous 16B halves at element offsets o0 / o1
__device__ __forceinline__ v16bf load_frag(const unsigned short* p, int o0, int o1) {
    BF16x16 u;
    u.h[0] = *(const v8bf*)(p + o0);
    u.h[1] = *(const v8bf*)(p + o1);
    return u.v;
}
__device__ __forceinline__ v8f wmma_bf16(v16bf a, v16bf b, v8f c) {
    // D(16x16 f32) = A(16x32 bf16) * B(32x16 bf16) + C
    return __builtin_amdgcn_wmma_f32_16x16x32_bf16(false, a, false, b, (short)0, c,
                                                   false, false);
}

// ---------- TDM (Tensor Data Mover) descriptor builders ----------
// Group1: data_size=2B, 2D tile 64(k) x 128(rows), row stride = K elements.
__device__ __forceinline__ v8i tdm_g1(unsigned K_, unsigned M_) {
    v8i g;
    g[0] = (int)(1u << 16);                                    // data_size=2B, wg_mask=0
    g[1] = (int)((K_ & 0xFFFFu) << 16);                        // tensor_dim0 lo16
    g[2] = (int)((K_ >> 16) | ((M_ & 0xFFFFu) << 16));         // dim0 hi | dim1 lo
    g[3] = (int)((M_ >> 16) | (64u << 16));                    // dim1 hi | tile_dim0=64
    g[4] = (int)(128u);                                        // tile_dim1=128, tile_dim2=0
    g[5] = (int)(K_);                                          // tensor_dim0_stride lo32
    g[6] = 0;                                                  // stride0 hi | stride1 lo
    g[7] = 0;                                                  // stride1 hi
    return g;
}
// Group0: count=1, lds byte address, 57-bit global tile address, type=2 (image)
__device__ __forceinline__ v4u tdm_g0(unsigned lds_byte, unsigned long long ga) {
    v4u g;
    g[0] = 1u;
    g[1] = lds_byte;
    g[2] = (unsigned)(ga & 0xFFFFFFFFu);
    g[3] = (unsigned)((ga >> 32) & 0x1FFFFFFu) | (2u << 30);
    return g;
}
__device__ __forceinline__ void tdm_load(v4u g0, v8i g1) {
    v4i z4 = {0, 0, 0, 0};
#if __clang_major__ >= 23
    v8i z8 = {0, 0, 0, 0, 0, 0, 0, 0};
    __builtin_amdgcn_tensor_load_to_lds(g0, g1, z4, z4, z8, 0);
#else
    __builtin_amdgcn_tensor_load_to_lds(g0, g1, z4, z4, 0);
#endif
}

// ---------- f32 -> bf16 convert ----------
__global__ __launch_bounds__(256) void cvt_f32_bf16(const float* __restrict__ in,
                                                    unsigned short* __restrict__ out,
                                                    int n) {
    int i = blockIdx.x * 256 + threadIdx.x;
    int stride = gridDim.x * 256;
    for (; i < n; i += stride) out[i] = f2bf(in[i]);
}

// ---------- LayerNorm (one row of C=768 per block), bf16 output ----------
__global__ __launch_bounds__(256) void layernorm_bf16(const float* __restrict__ x,
                                                      const float* __restrict__ w,
                                                      const float* __restrict__ b,
                                                      unsigned short* __restrict__ out) {
    const int row = blockIdx.x;
    const float* xr = x + (size_t)row * CC;
    float s = 0.f, s2 = 0.f;
    for (int i = threadIdx.x; i < CC; i += 256) { float v = xr[i]; s += v; s2 += v * v; }
    __shared__ float shs[256], shq[256];
    shs[threadIdx.x] = s; shq[threadIdx.x] = s2;
    __syncthreads();
    for (int off = 128; off > 0; off >>= 1) {
        if (threadIdx.x < off) {
            shs[threadIdx.x] += shs[threadIdx.x + off];
            shq[threadIdx.x] += shq[threadIdx.x + off];
        }
        __syncthreads();
    }
    const float mean = shs[0] * (1.0f / CC);
    const float var  = shq[0] * (1.0f / CC) - mean * mean;
    const float inv  = rsqrtf(var + 1e-5f);
    unsigned short* orow = out + (size_t)row * CC;
    for (int i = threadIdx.x; i < CC; i += 256)
        orow[i] = f2bf((xr[i] - mean) * inv * w[i] + b[i]);
}

// ---------- bf16 GEMM: out = A[M,K] @ B[K,N] + bias ----------
// MODE: 0 = bf16 out; 1 = bf16 out + relu; 2 = f32 out + f32 residual.
// Block: 256 threads = 8 waves, tile 128(M) x 128(N), K-step 64.
// Wave = 32x64 micro-tile (2x4 WMMA tiles).  A tile staged by TDM, B transposed by VALU.
template <int MODE>
__global__ __launch_bounds__(256) void gemm_bf16(
    const unsigned short* __restrict__ A,
    const unsigned short* __restrict__ Bw,
    const float* __restrict__ bias,
    const float* __restrict__ residual,   // MODE==2
    float* __restrict__ outF,             // MODE==2
    unsigned short* __restrict__ outH,    // MODE 0/1
    int M, int N, int K) {
    __shared__ unsigned short a_s[128 * 64];  // [row][k]          16 KB
    __shared__ unsigned short b_s[128 * 64];  // [n][k] transposed 16 KB

    const int tid   = threadIdx.x;
    const int lane  = tid & 31;
    const int wave  = tid >> 5;
    const int wm    = wave & 3;   // 4 waves over M (32 rows each)
    const int wn    = wave >> 2;  // 2 waves over N (64 cols each)
    const int mbase = blockIdx.y * 128;
    const int nbase = blockIdx.x * 128;

    v8f acc[2][4];
#pragma unroll
    for (int mt = 0; mt < 2; ++mt)
#pragma unroll
        for (int nt = 0; nt < 4; ++nt) acc[mt][nt] = zero8();

    // B staging indices: each thread owns one n-column, 32 k values
    const int bn   = tid & 127;
    const int kseg = (tid >> 7) * 32;  // 0 or 32
    const unsigned short* bptr = Bw + (size_t)kseg * N + (nbase + bn);

    // TDM descriptor (A tile): constant group1, per-iter group0
    const v8i g1 = tdm_g1((unsigned)K, (unsigned)M);
    const unsigned lds_a = (unsigned)(unsigned long long)(uintptr_t)(void*)a_s;
    const unsigned short* atile = A + (size_t)mbase * K;

    // fragment half-offsets (contiguous k order: lane half h holds k 16h..16h+15)
    const int halfl = lane & 15;
    const int ao0 = (lane < 16) ? 0 : 16, ao1 = (lane < 16) ? 8 : 24;   // A frag
    const int bo0 = (lane < 16) ? 0 : 16, bo1 = (lane < 16) ? 8 : 24;   // B frag

    for (int k0 = 0; k0 < K; k0 += 64) {
        // ---- stage A tile [128][64] via Tensor Data Mover (wave 0 issues) ----
        if (wave == 0) {
            tdm_load(tdm_g0(lds_a, (unsigned long long)(uintptr_t)(atile + k0)), g1);
        }
        // ---- stage B tile transposed [n][k] (all 256 threads) ----
#pragma unroll
        for (int j = 0; j < 32; ++j)
            b_s[bn * 64 + kseg + j] = bptr[(size_t)j * N];
        if (k0 + 64 < K)
            __builtin_prefetch(bptr + (size_t)64 * N, 0, 1);   // global_prefetch_b8
        if (wave == 0) __builtin_amdgcn_s_wait_tensorcnt(0);
        __syncthreads();

        // ---- 2 K-chunks of WMMA per stage ----
#pragma unroll
        for (int c = 0; c < 2; ++c) {
            v16bf af[2], bf[4];
#pragma unroll
            for (int mt = 0; mt < 2; ++mt)
                af[mt] = load_frag(&a_s[(wm * 32 + mt * 16 + halfl) * 64 + c * 32],
                                   ao0, ao1);
#pragma unroll
            for (int nt = 0; nt < 4; ++nt)
                bf[nt] = load_frag(&b_s[(wn * 64 + nt * 16 + halfl) * 64 + c * 32],
                                   bo0, bo1);
#pragma unroll
            for (int mt = 0; mt < 2; ++mt)
#pragma unroll
                for (int nt = 0; nt < 4; ++nt)
                    acc[mt][nt] = wmma_bf16(af[mt], bf[nt], acc[mt][nt]);
        }
        __syncthreads();
        bptr += (size_t)64 * N;
    }

    // ---- epilogue (branch-free, compile-time specialized) ----
    const int roff = (lane < 16) ? 0 : 8;
#pragma unroll
    for (int mt = 0; mt < 2; ++mt) {
#pragma unroll
        for (int nt = 0; nt < 4; ++nt) {
            const int col = nbase + wn * 64 + nt * 16 + halfl;
            const float bv = bias[col];
#pragma unroll
            for (int r = 0; r < 8; ++r) {
                const int row = mbase + wm * 32 + mt * 16 + roff + r;
                const size_t idx = (size_t)row * N + col;
                float v = acc[mt][nt][r] + bv;
                if (MODE == 1) v = fmaxf(v, 0.0f);
                if (MODE == 2) {
                    outF[idx] = v + residual[idx];
                } else {
                    outH[idx] = f2bf(v);
                }
            }
        }
    }
}

// ---------- Flash attention: causal, scale 1/8, bf16 in/out ----------
// Block: 256 threads = 8 waves; each wave owns a 16-row Q block.
__global__ __launch_bounds__(256) void attn_flash(
    const unsigned short* __restrict__ qkv,  // [B,T,3C] bf16
    unsigned short* __restrict__ o) {        // [B,T,C]  bf16
    __shared__ unsigned short v_s[8][32 * 64];  // per-wave V block [k][d]  32 KB
    __shared__ unsigned short p_s[8][16 * 32];  // per-wave P bounce [m][k]  8 KB

    const int lane = threadIdx.x & 31;
    const int wave = threadIdx.x >> 5;
    const int bh   = blockIdx.y;
    const int b    = bh / HH;
    const int h    = bh % HH;
    const int qb   = blockIdx.x * 128 + wave * 16;   // wave-uniform

    unsigned short* vs = v_s[wave];
    unsigned short* ps = p_s[wave];

    const unsigned short* basep = qkv + (size_t)b * TT * QKVLD + h * DD;
    const unsigned short* qp = basep;
    const unsigned short* kp = basep + CC;
    const unsigned short* vp = basep + 2 * CC;

    const int halfl = lane & 15;
    const int ao0 = (lane < 16) ? 0 : 16, ao1 = (lane < 16) ? 8 : 24;
    const int bo0 = (lane < 16) ? 0 : 16, bo1 = (lane < 16) ? 8 : 24;
    const int roff = (lane < 16) ? 0 : 8;

    // Q fragments (16 rows x 64, two Kc=32 chunks)
    v16bf qf[2];
    {
        const unsigned short* qrow = qp + (size_t)(qb + halfl) * QKVLD;
        qf[0] = load_frag(qrow,      ao0, ao1);
        qf[1] = load_frag(qrow + 32, ao0, ao1);
    }

    v8f oacc[4];
#pragma unroll
    for (int dt = 0; dt < 4; ++dt) oacc[dt] = zero8();
    float mr[8], lr[8];
#pragma unroll
    for (int r = 0; r < 8; ++r) { mr[r] = -3.0e38f; lr[r] = 0.0f; }

    const int nk = (qb + 47) >> 5;  // #32-wide K blocks covering [0, qb+16)
    for (int kb = 0; kb < nk; ++kb) {
        const int kbase = kb * 32;
        // stage V block [32][64] (one row per lane, 128B)
        {
            const uint4* src = (const uint4*)(vp + (size_t)(kbase + lane) * QKVLD);
            uint4* dst = (uint4*)&vs[lane * 64];
#pragma unroll
            for (int j = 0; j < 8; ++j) dst[j] = src[j];
        }
        // S = Q @ K^T : B-fragments load straight from bf16 K rows (contiguous)
        v8f s0 = zero8(), s1 = zero8();
        {
            const unsigned short* kr0 = kp + (size_t)(kbase + halfl) * QKVLD;
            const unsigned short* kr1 = kp + (size_t)(kbase + 16 + halfl) * QKVLD;
            s0 = wmma_bf16(qf[0], load_frag(kr0,      bo0, bo1), s0);
            s0 = wmma_bf16(qf[1], load_frag(kr0 + 32, bo0, bo1), s0);
            s1 = wmma_bf16(qf[0], load_frag(kr1,      bo0, bo1), s1);
            s1 = wmma_bf16(qf[1], load_frag(kr1 + 32, bo0, bo1), s1);
        }
        // online softmax (row stats reduce across 16-lane halves)
        const int kg0 = kbase + halfl;
        const int kg1 = kg0 + 16;
#pragma unroll
        for (int r = 0; r < 8; ++r) {
            const int qg = qb + roff + r;
            float a0 = s0[r] * 0.125f;
            float a1 = s1[r] * 0.125f;
            if (kg0 > qg) a0 = -3.0e38f;
            if (kg1 > qg) a1 = -3.0e38f;
            float mx = fmaxf(a0, a1);
#pragma unroll
            for (int off = 8; off >= 1; off >>= 1)
                mx = fmaxf(mx, __shfl_xor(mx, off, 32));
            const float mnew = fmaxf(mr[r], mx);
            const float p0 = __expf(a0 - mnew);
            const float p1 = __expf(a1 - mnew);
            float rs = p0 + p1;
#pragma unroll
            for (int off = 8; off >= 1; off >>= 1)
                rs += __shfl_xor(rs, off, 32);
            const float alpha = __expf(mr[r] - mnew);
            lr[r] = lr[r] * alpha + rs;
            mr[r] = mnew;
#pragma unroll
            for (int dt = 0; dt < 4; ++dt) oacc[dt][r] *= alpha;
            // C-layout -> LDS (row-major P block [16][32]) for A-fragment reload
            ps[(roff + r) * 32 + halfl]      = f2bf(p0);
            ps[(roff + r) * 32 + 16 + halfl] = f2bf(p1);
        }
        // O += P @ V   (P A-fragment via LDS bounce; V columns from LDS)
        const v16bf pf = load_frag(&ps[halfl * 32], ao0, ao1);
#pragma unroll
        for (int dt = 0; dt < 4; ++dt) {
            v16bf vf;
            const int col = dt * 16 + halfl;
            const int kk0 = (lane < 16) ? 0 : 16;
#pragma unroll
            for (int j = 0; j < 16; ++j)
                vf[j] = us2bf(vs[(kk0 + j) * 64 + col]);
            oacc[dt] = wmma_bf16(pf, vf, oacc[dt]);
        }
    }
    // write O / l  -> o[b, t, h*64 + d] (bf16)
    unsigned short* op = o + (size_t)b * TT * CC + h * DD;
#pragma unroll
    for (int r = 0; r < 8; ++r) {
        const float inv = 1.0f / lr[r];
        unsigned short* orow = op + (size_t)(qb + roff + r) * CC;
#pragma unroll
        for (int dt = 0; dt < 4; ++dt)
            orow[dt * 16 + halfl] = f2bf(oacc[dt][r] * inv);
    }
}

// ---------- host orchestration ----------
extern "C" void kernel_launch(void* const* d_in, const int* in_sizes, int n_in,
                              void* d_out, int out_size, void* d_ws, size_t ws_size,
                              hipStream_t stream) {
    (void)in_sizes; (void)n_in; (void)out_size; (void)ws_size;
    const float* x    = (const float*)d_in[0];
    const float* ln1w = (const float*)d_in[1];
    const float* ln1b = (const float*)d_in[2];
    const float* qkvw = (const float*)d_in[3];
    const float* qkvb = (const float*)d_in[4];
    const float* outw = (const float*)d_in[5];
    const float* outb = (const float*)d_in[6];
    const float* ln2w = (const float*)d_in[7];
    const float* ln2b = (const float*)d_in[8];
    const float* ff1w = (const float*)d_in[9];
    const float* ff1b = (const float*)d_in[10];
    const float* ff2w = (const float*)d_in[11];
    const float* ff2b = (const float*)d_in[12];
    float* out = (float*)d_out;

    char* ws = (char*)d_ws;
    auto alloc = [&](size_t bytes) -> void* {
        void* p = ws;
        ws += (bytes + 255) & ~(size_t)255;
        return p;
    };
    unsigned short* h1   = (unsigned short*)alloc((size_t)ROWS * CC * 2);
    unsigned short* qkvh = (unsigned short*)alloc((size_t)ROWS * QKVLD * 2);
    unsigned short* oh   = (unsigned short*)alloc((size_t)ROWS * CC * 2);
    float*          x2   = (float*)         alloc((size_t)ROWS * CC * 4);
    unsigned short* h2   = (unsigned short*)alloc((size_t)ROWS * CC * 2);
    unsigned short* a1   = (unsigned short*)alloc((size_t)ROWS * FFD * 2);
    unsigned short* wqkv = (unsigned short*)alloc((size_t)CC * QKVLD * 2);
    unsigned short* wout = (unsigned short*)alloc((size_t)CC * CC * 2);
    unsigned short* wff1 = (unsigned short*)alloc((size_t)CC * FFD * 2);
    unsigned short* wff2 = (unsigned short*)alloc((size_t)FFD * CC * 2);

    // weight conversion to bf16
    cvt_f32_bf16<<<512, 256, 0, stream>>>(qkvw, wqkv, CC * QKVLD);
    cvt_f32_bf16<<<512, 256, 0, stream>>>(outw, wout, CC * CC);
    cvt_f32_bf16<<<512, 256, 0, stream>>>(ff1w, wff1, CC * FFD);
    cvt_f32_bf16<<<512, 256, 0, stream>>>(ff2w, wff2, FFD * CC);

    // h1 = LN1(x)
    layernorm_bf16<<<ROWS, 256, 0, stream>>>(x, ln1w, ln1b, h1);
    // qkv = h1 @ qkv_w + qkv_b   (bf16 out)
    gemm_bf16<0><<<dim3(QKVLD / 128, ROWS / 128), 256, 0, stream>>>(
        h1, wqkv, qkvb, nullptr, nullptr, qkvh, ROWS, QKVLD, CC);
    // o = causal softmax(q k^T / 8) v   (bf16 out)
    attn_flash<<<dim3(TT / 128, BB * HH), 256, 0, stream>>>(qkvh, oh);
    // x2 = x + o @ out_w + out_b   (f32)
    gemm_bf16<2><<<dim3(CC / 128, ROWS / 128), 256, 0, stream>>>(
        oh, wout, outb, x, x2, nullptr, ROWS, CC, CC);
    // h2 = LN2(x2)
    layernorm_bf16<<<ROWS, 256, 0, stream>>>(x2, ln2w, ln2b, h2);
    // a1 = relu(h2 @ ff1_w + ff1_b)   (bf16)
    gemm_bf16<1><<<dim3(FFD / 128, ROWS / 128), 256, 0, stream>>>(
        h2, wff1, ff1b, nullptr, nullptr, a1, ROWS, FFD, CC);
    // out = x2 + a1 @ ff2_w + ff2_b   (f32 -> d_out)
    gemm_bf16<2><<<dim3(CC / 128, ROWS / 128), 256, 0, stream>>>(
        a1, wff2, ff2b, x2, out, nullptr, ROWS, CC, FFD);
}
